// QuantEdgeTransition_44555990729114
// MI455X (gfx1250) — matrix operand order repaired and checked by reading, not compiled, not run on hardware
//
#include <hip/hip_runtime.h>
#include <hip/hip_bf16.h>
#include <cstdint>

// ---------------------------------------------------------------------------
// CDNA5 (gfx1250) edge-transition MLP.
// node: [384,256] -> proj [384,128] (prep kernel, bf16)
// x row = [edge(128) | node_i(128) | node_j(128)]  (HID=384)
// h1 = relu(x@W1+b1); h2 = relu(h1@W2+b2)+x; y = h2@Wf+bf; LN(128)
// GEMMs via v_wmma_f32_16x16x32_bf16 (wave32). Weight k-slices are staged
// into LDS by the Tensor Data Mover (tensor_load_to_lds), double-buffered so
// the DMA overlaps the WMMA compute; tracked with TENSORcnt.
// ---------------------------------------------------------------------------

typedef __attribute__((ext_vector_type(16))) __bf16 v16bf;
typedef __attribute__((ext_vector_type(8)))  float  v8f;
typedef __attribute__((ext_vector_type(4)))  unsigned int u32x4;
typedef __attribute__((ext_vector_type(8)))  unsigned int u32x8;

#define NPTS 384
#define HID  384
#define DE   128
#define DB   128
#define MT   64            // pair-rows per block
#define XS_STRIDE 392      // bf16 elems/row (384 + 8 pad) -> 784B, 16B aligned
#define WK_STRIDE 40       // bf16 elems/row (32 + 8 pad)  -> 80B, 16B aligned
#define LDS_WK0    0       // 384*80 = 30720 B  (TDM dest, buffer 0)
#define LDS_WK1    30720   // 384*80 = 30720 B  (TDM dest, buffer 1)
#define LDS_XS_OFF 61440   // 64*392*2 = 50176 B
#define LDS_HS_OFF 111616  // 64*392*2 = 50176 B
#define LDS_TOTAL  161792  // ~158 KB

static __device__ __forceinline__ unsigned short f2bf(float f) {
    union { float f; uint32_t u; } c; c.f = f;
    uint32_t u = c.u + 0x7FFFu + ((c.u >> 16) & 1u);   // round-nearest-even
    return (unsigned short)(u >> 16);
}
static __device__ __forceinline__ float bf2f(unsigned short h) {
    union { uint32_t u; float f; } c; c.u = ((uint32_t)h) << 16; return c.f;
}
static __device__ __forceinline__ v16bf as_v16bf(uint4 a, uint4 b) {
    union { uint4 q[2]; v16bf v; } u; u.q[0] = a; u.q[1] = b; return u.v;
}

// ---- Tensor Data Mover: DMA one 32 x nrows bf16 weight slice into LDS -----
// Source: WT (row-major, nrows rows of HID bf16), tile = rows [0,nrows) x
// k [kslice*32, +32). LDS dest gets 16B padding after each 64B row
// (pad_interval code 3 = 16 DWORDs, pad_amount code 3 = 4 DWORDs) so the
// staged layout matches WK_STRIDE = 40 bf16 elems.
static __device__ __forceinline__ void tdm_issue(const void* gsrc, int nrows,
                                                 int kslice, uint32_t lds_addr) {
    uint64_t ga = (uint64_t)(uintptr_t)gsrc + (uint64_t)kslice * 64u;
    u32x4 g0 = { 1u,                                   // count=1, user desc
                 lds_addr,                             // lds_addr (bytes)
                 (uint32_t)ga,                         // global_addr[31:0]
                 (((uint32_t)(ga >> 32)) & 0x01FFFFFFu) | 0x80000000u }; // type=2
    u32x8 g1 = { 0x06D10000u,                          // ds=2B,pad_en,intv=3,amt=3
                 ((uint32_t)HID) << 16,                // tensor_dim0 = 384
                 ((uint32_t)nrows) << 16,              // tensor_dim1 = nrows
                 32u << 16,                            // tile_dim0 = 32
                 (uint32_t)nrows,                      // tile_dim1 = nrows
                 (uint32_t)HID,                        // tensor_dim0_stride = 384
                 0u, 0u };
    u32x4 g2 = { 0u, 0u, 0u, 0u };                     // dims 2/3 unused
    u32x4 g3 = { 0u, 0u, 0u, 0u };
    asm volatile("tensor_load_to_lds %0, %1, %2, %3"
                 :: "s"(g0), "s"(g1), "s"(g2), "s"(g3)
                 : "memory");
}

// ------------------------------- prep kernels ------------------------------

// WT[n*K + k] = bf16(W[k*Ncol + n])   (W is [K, Ncol] row-major)
__global__ void transpose_to_bf16(const float* __restrict__ W,
                                  unsigned short* __restrict__ WT,
                                  int K, int Ncol) {
    int idx = blockIdx.x * blockDim.x + threadIdx.x;
    if (idx >= K * Ncol) return;
    int n = idx / K;
    int k = idx % K;
    WT[n * K + k] = f2bf(W[k * Ncol + n]);
}

// node_bf[i*128+d] = bf16(node_embed[i,:] @ Wi[:,d] + bi[d])
__global__ void node_project(const float* __restrict__ node_embed,
                             const float* __restrict__ Wi,
                             const float* __restrict__ bi,
                             unsigned short* __restrict__ node_bf) {
    int idx = blockIdx.x * blockDim.x + threadIdx.x;
    if (idx >= NPTS * DB) return;
    int i = idx / DB, d = idx % DB;
    float acc = bi[d];
    const float* row = node_embed + i * 256;
    for (int k = 0; k < 256; ++k) acc += row[k] * Wi[k * DB + d];
    node_bf[idx] = f2bf(acc);
}

// ------------------------- shared GEMM inner loop --------------------------
// Input tile: bf16 in LDS at byte offset in_off, stride XS_STRIDE, 64 rows.
// Weights: bf16 WT[n][k] in global (rows = ncols, row length = HID), staged
// per 32-wide k-slice into double-buffered LDS by the TDM (wave 0 issues).
static __device__ __forceinline__ void mm_tile(
    char* smem, const unsigned short* __restrict__ wt, int ncols,
    int in_off, v8f* acc, int nt, int colbase, int tid, int lane)
{
    const int half = lane >> 4;
    const int m    = lane & 15;
    const int rw   = ((tid >> 5) & 3) * 16;
    const uint32_t lds_base = (uint32_t)(uintptr_t)smem;

    if (tid < 32) tdm_issue(wt, ncols, 0, lds_base + LDS_WK0);   // prologue

    for (int kt = 0; kt < 12; ++kt) {
        if (tid < 32) __builtin_amdgcn_s_wait_tensorcnt(0);      // slice kt ready
        __syncthreads();                                         // publish; prev reads done
        if (tid < 32 && kt < 11)                                 // prefetch kt+1,
            tdm_issue(wt, ncols, kt + 1,                         // overlaps compute
                      lds_base + (((kt + 1) & 1) ? LDS_WK1 : LDS_WK0));
        const int wk = (kt & 1) ? LDS_WK1 : LDS_WK0;
        // A fragment (16x32 bf16, ISA layout): lanes 0-15 K={0..7,16..23},
        // lanes 16-31 K={8..15,24..31}; two 16B LDS reads per lane.
        const char* abase = smem + in_off +
            (size_t)((rw + m) * XS_STRIDE + kt * 32 + half * 8) * 2;
        uint4 a0 = *(const uint4*)abase;
        uint4 a1 = *(const uint4*)(abase + 32);
        v16bf av = as_v16bf(a0, a1);
        for (int ct = 0; ct < nt; ++ct) {
            int n0 = colbase + ct * 16;
            // B fragment (32x16): lane n = lane%16, lanes 0-15 K={0..7,16..23}, lanes 16-31 K={8..15,24..31}
            const char* bbase = smem + wk +
                (size_t)(n0 + m) * (WK_STRIDE * 2) + half * 16;
            uint4 b0 = *(const uint4*)bbase;
            uint4 b1 = *(const uint4*)(bbase + 32);
            v16bf bv = as_v16bf(b0, b1);
            acc[ct] = __builtin_amdgcn_wmma_f32_16x16x32_bf16(
                false, av, false, bv, (short)0, acc[ct], false, false);
        }
    }
}

// ------------------------------- main kernel -------------------------------

__global__ __launch_bounds__(256, 1)
void edge_transition_kernel(const float* __restrict__ edge_embed,
                            const unsigned short* __restrict__ node_bf,
                            const unsigned short* __restrict__ wt1,
                            const unsigned short* __restrict__ wt2,
                            const unsigned short* __restrict__ wtf,
                            const float* __restrict__ b1,
                            const float* __restrict__ b2,
                            const float* __restrict__ bfin,
                            const float* __restrict__ gamma,
                            const float* __restrict__ beta,
                            float* __restrict__ out)
{
    extern __shared__ char smem[];
    const int tid  = threadIdx.x;
    const int lane = tid & 31;
    const int wave = tid >> 5;
    const int half = lane >> 4;
    const int m    = lane & 15;
    const int rw   = (wave & 3) * 16;

    const int p0 = blockIdx.x * MT;     // 64 consecutive pairs, fixed i
    const int i  = p0 / NPTS;
    const int j0 = p0 % NPTS;

    unsigned short* Xs = (unsigned short*)(smem + LDS_XS_OFF);
    unsigned short* Hs = (unsigned short*)(smem + LDS_HS_OFF);

    // ---- build x tile (bf16) ----
    const float* erow = edge_embed + ((size_t)i * NPTS + j0) * DE;
    const unsigned short* nrow_i = node_bf + i * DB;
    for (int idx = tid; idx < MT * HID; idx += 256) {
        int r = idx / HID, c = idx % HID;
        unsigned short v;
        if (c < DE)       v = f2bf(erow[(size_t)r * DE + c]);
        else if (c < 256) v = nrow_i[c - DE];
        else              v = node_bf[(j0 + r) * DB + (c - 256)];
        Xs[r * XS_STRIDE + c] = v;
    }

    v8f acc[12];
    const int colbase = (wave >> 2) * 192;   // two column-halves of 192

    // ---- layer 1: h1 = relu(x@W1 + b1) ----
    for (int t = 0; t < 12; ++t) for (int e = 0; e < 8; ++e) acc[t][e] = 0.0f;
    mm_tile(smem, wt1, HID, LDS_XS_OFF, acc, 12, colbase, tid, lane);
    __syncthreads();
    for (int ct = 0; ct < 12; ++ct) {
        int n0 = colbase + ct * 16;
        float bb = b1[n0 + m];
        for (int r = 0; r < 8; ++r) {
            int mm = r + 8 * half;   // C/D layout: VGPR r -> M = r + 8*(lane/16)
            float v = acc[ct][r] + bb;
            v = v > 0.0f ? v : 0.0f;
            Hs[(rw + mm) * XS_STRIDE + n0 + m] = f2bf(v);
        }
    }

    // ---- layer 2: h = relu(h1@W2 + b2) + x   (in-place into Hs) ----
    for (int t = 0; t < 12; ++t) for (int e = 0; e < 8; ++e) acc[t][e] = 0.0f;
    mm_tile(smem, wt2, HID, LDS_HS_OFF, acc, 12, colbase, tid, lane);
    __syncthreads();   // all Hs reads finished before overwrite
    for (int ct = 0; ct < 12; ++ct) {
        int n0 = colbase + ct * 16;
        float bb = b2[n0 + m];
        for (int r = 0; r < 8; ++r) {
            int mm = r + 8 * half;
            float v = acc[ct][r] + bb;
            v = v > 0.0f ? v : 0.0f;
            v += bf2f(Xs[(rw + mm) * XS_STRIDE + n0 + m]);   // residual
            Hs[(rw + mm) * XS_STRIDE + n0 + m] = f2bf(v);
        }
    }

    // ---- layer 3: y = h@Wf + bf  (128 cols; 4 tiles per wave) ----
    const int colbase3 = (wave >> 2) * 64;
    for (int t = 0; t < 4; ++t) for (int e = 0; e < 8; ++e) acc[t][e] = 0.0f;
    mm_tile(smem, wtf, DE, LDS_HS_OFF, acc, 4, colbase3, tid, lane);
    __syncthreads();
    float* Ys = (float*)(smem + LDS_XS_OFF);   // x no longer needed
    for (int ct = 0; ct < 4; ++ct) {
        int n0 = colbase3 + ct * 16;
        float bb = bfin[n0 + m];
        for (int r = 0; r < 8; ++r) {
            int mm = r + 8 * half;
            Ys[(rw + mm) * 132 + n0 + m] = acc[ct][r] + bb;  // stride 132 -> 528B
        }
    }
    __syncthreads();

    // ---- layer norm over 128 channels: 4 lanes per row ----
    {
        int row = tid >> 2;                 // 0..63
        int sub = tid & 3;                  // 32 cols each
        const float* yr = Ys + row * 132 + sub * 32;
        float4 vals[8];
        float s = 0.0f, sq = 0.0f;
        for (int c = 0; c < 8; ++c) {
            float4 v = *(const float4*)(yr + c * 4);
            vals[c] = v;
            s  += v.x + v.y + v.z + v.w;
            sq += v.x * v.x + v.y * v.y + v.z * v.z + v.w * v.w;
        }
        s  += __shfl_xor(s, 1);  s  += __shfl_xor(s, 2);
        sq += __shfl_xor(sq, 1); sq += __shfl_xor(sq, 2);
        float mean = s * (1.0f / 128.0f);
        float var  = sq * (1.0f / 128.0f) - mean * mean;
        float rs   = rsqrtf(var + 1e-5f);
        float* orow = out + ((size_t)(p0 + row)) * DE + sub * 32;
        const float* g  = gamma + sub * 32;
        const float* bt = beta  + sub * 32;
        for (int c = 0; c < 8; ++c) {
            float4 v = vals[c];
            float4 o;
            o.x = (v.x - mean) * rs * g[c * 4 + 0] + bt[c * 4 + 0];
            o.y = (v.y - mean) * rs * g[c * 4 + 1] + bt[c * 4 + 1];
            o.z = (v.z - mean) * rs * g[c * 4 + 2] + bt[c * 4 + 2];
            o.w = (v.w - mean) * rs * g[c * 4 + 3] + bt[c * 4 + 3];
            *(float4*)(orow + c * 4) = o;
        }
    }
}

// ------------------------------- launcher ----------------------------------

extern "C" void kernel_launch(void* const* d_in, const int* in_sizes, int n_in,
                              void* d_out, int out_size, void* d_ws, size_t ws_size,
                              hipStream_t stream) {
    const float* node_embed = (const float*)d_in[0];
    const float* edge_embed = (const float*)d_in[1];
    const float* Wi    = (const float*)d_in[2];
    const float* bi    = (const float*)d_in[3];
    const float* W1    = (const float*)d_in[4];
    const float* b1    = (const float*)d_in[5];
    const float* W2    = (const float*)d_in[6];
    const float* b2    = (const float*)d_in[7];
    const float* Wf    = (const float*)d_in[8];
    const float* bff   = (const float*)d_in[9];
    const float* gamma = (const float*)d_in[10];
    const float* beta  = (const float*)d_in[11];
    float* out = (float*)d_out;

    // workspace (bf16): WT1 | WT2 | WfT | node_proj   (~786 KB total)
    unsigned short* wt1     = (unsigned short*)d_ws;
    unsigned short* wt2     = wt1 + 384 * 384;
    unsigned short* wtf     = wt2 + 384 * 384;
    unsigned short* node_bf = wtf + 128 * 384;

    transpose_to_bf16<<<(384 * 384 + 255) / 256, 256, 0, stream>>>(W1, wt1, 384, 384);
    transpose_to_bf16<<<(384 * 384 + 255) / 256, 256, 0, stream>>>(W2, wt2, 384, 384);
    transpose_to_bf16<<<(128 * 384 + 255) / 256, 256, 0, stream>>>(Wf, wtf, 384, 128);
    node_project<<<(384 * 128 + 255) / 256, 256, 0, stream>>>(node_embed, Wi, bi, node_bf);

    dim3 grid(NPTS * NPTS / MT);   // 2304 blocks
    edge_transition_kernel<<<grid, 256, LDS_TOTAL, stream>>>(
        edge_embed, node_bf, wt1, wt2, wtf, b1, b2, bff, gamma, beta, out);
}
